// EGNN_50792283242914
// MI455X (gfx1250) — compile-verified, matrix operand order checked by reading, not run on hardware
//
#include <hip/hip_runtime.h>

// EGNN forward for MI455X (gfx1250, wave32, WMMA).
// Compute-bound (~465 GFLOPs); edge MLP fused so per-edge activations never
// touch HBM; h (25.6MB) is L2-resident so gathers are cheap. Matmuls run on
// v_wmma_f32_16x16x32_f16 with f32 accumulate; weights pre-converted to f16,
// stored transposed [col][k] (K padded to mult. of 32) so fragments load as
// two ds_load_b128 per lane. A-fragments are hoisted into registers once per
// stage and reused across all 8 N-tiles (8x less A-side LDS traffic).

#define NN   50000
#define NE   800000
#define NL   4
#define HD   128
#define CDD  3
#define VDD  3
#define FDD  16
#define EDD  16
#define DIN  273      // 2H + ED + 1
#define DINP 288      // padded to multiple of 32
#define NROW 22       // CD+VD+FD

typedef __attribute__((ext_vector_type(16))) _Float16 v16h;
typedef __attribute__((ext_vector_type(8)))  _Float16 v8h;
typedef __attribute__((ext_vector_type(4)))  _Float16 v4h;
typedef __attribute__((ext_vector_type(8)))  float    v8f;

__device__ __forceinline__ float silu_f(float x) { return x / (1.0f + __expf(-x)); }

// Load one 16x32 f16 fragment (A or B operand) from LDS, per CDNA5 ISA layout:
// lane L holds row/col (L&15), k-base (L>>4)*8; halves 0..7 -> k..k+7,
// halves 8..15 -> k+16..k+23. Two ds_load_b128 per lane.
__device__ __forceinline__ v16h ld_frag(const _Float16* p) {
  v8h lo = *(const v8h*)(p);
  v8h hi = *(const v8h*)(p + 16);
  v16h r;
#pragma unroll
  for (int i = 0; i < 8; ++i) { r[i] = lo[i]; r[i + 8] = hi[i]; }
  return r;
}

// One MLP stage: out[16 x nnt*16] = A(16xK) @ W(K x nnt*16), K = NKC*32.
// A fragments loaded once into registers, reused across all N-tiles; only the
// B fragment streams from LDS per WMMA. epi(nt, acc) handles bias/act/store.
template <int NKC, typename EPI>
__device__ __forceinline__ void mm_stage(const _Float16* at, int astr,
                                         const _Float16* wt, int wstr,
                                         int nnt, int lane, EPI&& epi) {
  const int kb = (lane >> 4) << 3;
  const _Float16* ap = at + (lane & 15) * astr + kb;
  v16h afrag[NKC];
#pragma unroll
  for (int kc = 0; kc < NKC; ++kc) afrag[kc] = ld_frag(ap + kc * 32);
  for (int nt = 0; nt < nnt; ++nt) {
    const _Float16* bp = wt + (nt * 16 + (lane & 15)) * wstr + kb;
    v8f acc = {0.f, 0.f, 0.f, 0.f, 0.f, 0.f, 0.f, 0.f};
#pragma unroll
    for (int kc = 0; kc < NKC; ++kc) {
      v16h b = ld_frag(bp + kc * 32);
      acc = __builtin_amdgcn_wmma_f32_16x16x32_f16(false, afrag[kc], false, b,
                                                   (short)0, acc, false, false);
    }
    epi(nt, acc);
  }
}

// ---- weight prep: f32 [K][Nc] -> f16 [Np][Kp] (transposed, zero padded) ----
__global__ void egnn_wprep(const float* __restrict__ src, _Float16* __restrict__ dst,
                           int K, int Nc, int Kp, int Np) {
  int i = blockIdx.x * blockDim.x + threadIdx.x;
  if (i >= Np * Kp) return;
  int n = i / Kp, k = i % Kp;
  float v = (n < Nc && k < K) ? src[k * Nc + n] : 0.0f;
  dst[i] = (_Float16)v;
}

// ---- embedding + coords/vels extraction: h0 = feat @ emb_w + emb_b ----
__global__ void egnn_embed(const float* __restrict__ nodes,
                           const float* __restrict__ emb_w,
                           const float* __restrict__ emb_b,
                           float* __restrict__ h, float* __restrict__ coords,
                           float* __restrict__ vels) {
  int i = blockIdx.x;        // node
  int j = threadIdx.x;       // 0..127
  __shared__ float f[FDD];
  if (j < FDD) f[j] = nodes[(long long)i * NROW + CDD + VDD + j];
  if (j < CDD) coords[i * CDD + j] = nodes[(long long)i * NROW + j];
  if (j < VDD) vels[i * VDD + j]   = nodes[(long long)i * NROW + CDD + j];
  __syncthreads();
  float s = emb_b[j];
#pragma unroll
  for (int k = 0; k < FDD; ++k) s += f[k] * emb_w[k * HD + j];
  h[(long long)i * HD + j] = s;
}

// ---- segment counts ----
__global__ void egnn_cnt(const long long* __restrict__ eidx, float* __restrict__ cnt) {
  int e = blockIdx.x * blockDim.x + threadIdx.x;
  if (e < NE) atomicAdd(&cnt[(int)eidx[e]], 1.0f);
}

// ---- fused edge pipeline ----
// x = [h[s], h[e], nrm, ef]  (273 -> 288)
// t1 = silu(x @ ew1 + eb1); m = silu(t1 @ ew2 + eb2)
// t2 = silu(m @ cw1 + cb1); sc = t2 @ cw2 + cb2
// atomics: agg[s] += m ; cacc[s] += cd * sc
#define WTE (HD*DINP + HD*HD + HD*HD + 16*HD)   // 71680 halves
__global__ __launch_bounds__(256) void egnn_edge(
    const float* __restrict__ h, const float* __restrict__ coords,
    const long long* __restrict__ eidx, const float* __restrict__ efeat,
    const _Float16* __restrict__ wblk,
    const float* __restrict__ eb1, const float* __restrict__ eb2,
    const float* __restrict__ cb1, const float* __restrict__ cb2,
    float* __restrict__ agg, float* __restrict__ cacc) {
  extern __shared__ char smem_raw[];
  _Float16* sw    = (_Float16*)smem_raw;
  _Float16* s_ew1 = sw;
  _Float16* s_ew2 = s_ew1 + HD * DINP;
  _Float16* s_cw1 = s_ew2 + HD * HD;
  _Float16* s_cw2 = s_cw1 + HD * HD;
  _Float16* tiles = sw + WTE;                          // 8 waves * (16*DINP + 16*HD)
  int*      meta  = (int*)(tiles + 8 * (16 * DINP + 16 * HD));

  const int wave = threadIdx.x >> 5, lane = threadIdx.x & 31;
  const int lcol = lane & 15;
  const int mbase = (lane >> 4) << 3;
  _Float16* xt = tiles + wave * (16 * DINP + 16 * HD);
  _Float16* yt = xt + 16 * DINP;
  int*   estart = meta + wave * (16 + 16 + 48);
  int*   eend   = estart + 16;
  float* cdm    = (float*)(eend + 16);

  // cooperative weight stage (f16, ~143 KB)
  for (int i = threadIdx.x; i < WTE / 4; i += blockDim.x)
    ((v4h*)sw)[i] = ((const v4h*)wblk)[i];
  __syncthreads();

  const int ntiles = (NE + 15) >> 4;
  for (int t = blockIdx.x * 8 + wave; t < ntiles; t += gridDim.x * 8) {
    const int e0 = t << 4;
    if (lane < 16) {
      int e = e0 + lane; if (e >= NE) e = NE - 1;
      int s = (int)eidx[e], d = (int)eidx[NE + e];
      estart[lane] = s; eend[lane] = d;
      float cx = coords[s * 3 + 0] - coords[d * 3 + 0];
      float cy = coords[s * 3 + 1] - coords[d * 3 + 1];
      float cz = coords[s * 3 + 2] - coords[d * 3 + 2];
      cdm[lane * 3 + 0] = cx; cdm[lane * 3 + 1] = cy; cdm[lane * 3 + 2] = cz;
      _Float16* xr = xt + lane * DINP;
      xr[2 * HD] = (_Float16)sqrtf(cx * cx + cy * cy + cz * cz);
#pragma unroll
      for (int k = 0; k < EDD; ++k)
        xr[2 * HD + 1 + k] = (_Float16)efeat[(long long)e * EDD + k];
#pragma unroll
      for (int k = DIN; k < DINP; ++k) xr[k] = (_Float16)0.f;
    }
    // gather h[start], h[end] (float4, L2-resident) -> f16 A tile
    for (int i = lane; i < 16 * 64; i += 32) {
      int er = i >> 6, q = i & 63, c4 = q & 31;
      int node = (q < 32) ? estart[er] : eend[er];
      const float4 v = ((const float4*)(h + (long long)node * HD))[c4];
      v4h hv = {(_Float16)v.x, (_Float16)v.y, (_Float16)v.z, (_Float16)v.w};
      *(v4h*)(xt + er * DINP + ((q < 32) ? 0 : HD) + c4 * 4) = hv;
    }
    // stage 1: t1 = silu(x @ ew1 + eb1) -> yt
    mm_stage<DINP / 32>(xt, DINP, s_ew1, DINP, 8, lane, [&](int nt, v8f acc) {
      float b = eb1[nt * 16 + lcol];
#pragma unroll
      for (int r = 0; r < 8; ++r)
        yt[(r + mbase) * HD + nt * 16 + lcol] = (_Float16)silu_f(acc[r] + b);
    });
    // stage 2: m = silu(t1 @ ew2 + eb2) -> xt cols 0..127 ; scatter agg
    mm_stage<HD / 32>(yt, HD, s_ew2, HD, 8, lane, [&](int nt, v8f acc) {
      float b = eb2[nt * 16 + lcol];
#pragma unroll
      for (int r = 0; r < 8; ++r) {
        int M = r + mbase;
        float mv = silu_f(acc[r] + b);
        xt[M * DINP + nt * 16 + lcol] = (_Float16)mv;
        if (e0 + M < NE)
          atomicAdd(&agg[(long long)estart[M] * HD + nt * 16 + lcol], mv);
      }
    });
    // stage 3: t2 = silu(m @ cw1 + cb1) -> yt
    mm_stage<HD / 32>(xt, DINP, s_cw1, HD, 8, lane, [&](int nt, v8f acc) {
      float b = cb1[nt * 16 + lcol];
#pragma unroll
      for (int r = 0; r < 8; ++r)
        yt[(r + mbase) * HD + nt * 16 + lcol] = (_Float16)silu_f(acc[r] + b);
    });
    // stage 4: sc = t2 @ cw2 + cb2 ; cacc[s] += cd * sc
    mm_stage<HD / 32>(yt, HD, s_cw2, HD, 1, lane, [&](int, v8f acc) {
      if (lcol < CDD) {
        float b = cb2[lcol];
#pragma unroll
        for (int r = 0; r < 8; ++r) {
          int M = r + mbase;
          if (e0 + M < NE)
            atomicAdd(&cacc[(long long)estart[M] * CDD + lcol],
                      cdm[M * 3 + lcol] * (acc[r] + b));
        }
      }
    });
  }
}

// ---- fused node pipeline ----
// vfac = silu(h @ vw1 + vb1) @ vw2 + vb2
// coords' = coords + cacc/cnt + vfac*vels
// h' = h + silu([h,agg] @ nw1 + nb1) @ nw2 + nb2
#define WTN (HD*HD + 16*HD + HD*2*HD + HD*HD)    // 67584 halves
__global__ __launch_bounds__(256) void egnn_node(
    const float* __restrict__ h, const float* __restrict__ agg,
    const float* __restrict__ coords, const float* __restrict__ cacc,
    const float* __restrict__ cnt, const float* __restrict__ vels,
    const _Float16* __restrict__ wblk,
    const float* __restrict__ vb1, const float* __restrict__ vb2,
    const float* __restrict__ nb1, const float* __restrict__ nb2,
    float* __restrict__ hout, float* __restrict__ coords_out) {
  extern __shared__ char smem_raw[];
  _Float16* sw    = (_Float16*)smem_raw;
  _Float16* s_vw1 = sw;
  _Float16* s_vw2 = s_vw1 + HD * HD;
  _Float16* s_nw1 = s_vw2 + 16 * HD;
  _Float16* s_nw2 = s_nw1 + HD * 2 * HD;
  _Float16* tiles = sw + WTN;
  float*    vfl   = (float*)(tiles + 8 * (16 * 2 * HD + 16 * HD));

  const int wave = threadIdx.x >> 5, lane = threadIdx.x & 31;
  const int lcol = lane & 15;
  const int mbase = (lane >> 4) << 3;
  _Float16* xn = tiles + wave * (16 * 2 * HD + 16 * HD);
  _Float16* yt = xn + 16 * 2 * HD;
  float* vfac = vfl + wave * 16;

  for (int i = threadIdx.x; i < WTN / 4; i += blockDim.x)
    ((v4h*)sw)[i] = ((const v4h*)wblk)[i];
  __syncthreads();

  const int ntiles = (NN + 15) >> 4;
  for (int t = blockIdx.x * 8 + wave; t < ntiles; t += gridDim.x * 8) {
    const int n0 = t << 4;
    // stage [h | agg] tile (16 x 256)
    for (int i = lane; i < 16 * 64; i += 32) {
      int nr = i >> 6, q = i & 63, c4 = q & 31;
      int node = n0 + nr; if (node >= NN) node = NN - 1;
      const float* src = (q < 32) ? (h + (long long)node * HD)
                                  : (agg + (long long)node * HD);
      const float4 v = ((const float4*)src)[c4];
      v4h hv = {(_Float16)v.x, (_Float16)v.y, (_Float16)v.z, (_Float16)v.w};
      *(v4h*)(xn + nr * (2 * HD) + ((q < 32) ? 0 : HD) + c4 * 4) = hv;
    }
    // vt = silu(h @ vw1 + vb1) -> yt
    mm_stage<HD / 32>(xn, 2 * HD, s_vw1, HD, 8, lane, [&](int nt, v8f acc) {
      float b = vb1[nt * 16 + lcol];
#pragma unroll
      for (int r = 0; r < 8; ++r)
        yt[(r + mbase) * HD + nt * 16 + lcol] = (_Float16)silu_f(acc[r] + b);
    });
    // vfac = vt @ vw2 + vb2 (only column 0 of the padded 16)
    mm_stage<HD / 32>(yt, HD, s_vw2, HD, 1, lane, [&](int, v8f acc) {
      if (lcol == 0) {
        float b = vb2[0];
#pragma unroll
        for (int r = 0; r < 8; ++r) vfac[r + mbase] = acc[r] + b;
      }
    });
    // coords update
    if (lane < 16) {
      int i = n0 + lane;
      if (i < NN) {
        float cn = fmaxf(cnt[i], 1.0f);
        float vf = vfac[lane];
#pragma unroll
        for (int d = 0; d < CDD; ++d)
          coords_out[i * CDD + d] = coords[i * CDD + d] +
                                    cacc[i * CDD + d] / cn +
                                    vf * vels[i * VDD + d];
      }
    }
    // u = silu([h,agg] @ nw1 + nb1) -> yt
    mm_stage<(2 * HD) / 32>(xn, 2 * HD, s_nw1, 2 * HD, 8, lane,
                            [&](int nt, v8f acc) {
      float b = nb1[nt * 16 + lcol];
#pragma unroll
      for (int r = 0; r < 8; ++r)
        yt[(r + mbase) * HD + nt * 16 + lcol] = (_Float16)silu_f(acc[r] + b);
    });
    // h' = h + u @ nw2 + nb2  (residual re-read in f32 for accuracy)
    mm_stage<HD / 32>(yt, HD, s_nw2, HD, 8, lane, [&](int nt, v8f acc) {
      int g = nt * 16 + lcol;
      float b = nb2[g];
#pragma unroll
      for (int r = 0; r < 8; ++r) {
        int i = n0 + r + mbase;
        if (i < NN)
          hout[(long long)i * HD + g] = h[(long long)i * HD + g] + acc[r] + b;
      }
    });
  }
}

extern "C" void kernel_launch(void* const* d_in, const int* in_sizes, int n_in,
                              void* d_out, int out_size, void* d_ws, size_t ws_size,
                              hipStream_t stream) {
  (void)in_sizes; (void)n_in; (void)out_size; (void)ws_size;
  const float*     nodes = (const float*)d_in[0];
  const long long* eidx  = (const long long*)d_in[1];   // reference: int64
  const float*     efeat = (const float*)d_in[2];
  const float*     emb_w = (const float*)d_in[3];
  const float*     emb_b = (const float*)d_in[4];
  const float* ew1 = (const float*)d_in[5];  const float* eb1 = (const float*)d_in[6];
  const float* ew2 = (const float*)d_in[7];  const float* eb2 = (const float*)d_in[8];
  const float* cw1 = (const float*)d_in[9];  const float* cb1 = (const float*)d_in[10];
  const float* cw2 = (const float*)d_in[11]; const float* cb2 = (const float*)d_in[12];
  const float* vw1 = (const float*)d_in[13]; const float* vb1 = (const float*)d_in[14];
  const float* vw2 = (const float*)d_in[15]; const float* vb2 = (const float*)d_in[16];
  const float* nw1 = (const float*)d_in[17]; const float* nb1 = (const float*)d_in[18];
  const float* nw2 = (const float*)d_in[19]; const float* nb2 = (const float*)d_in[20];

  // ---- workspace carve (~80.5 MB) ----
  char* ws = (char*)d_ws;
  size_t off = 0;
  auto carve = [&](size_t bytes) -> char* {
    char* p = ws + off; off += (bytes + 255) & ~(size_t)255; return p;
  };
  float* h_a  = (float*)carve((size_t)NN * HD * 4);
  float* h_b  = (float*)carve((size_t)NN * HD * 4);
  float* agg  = (float*)carve((size_t)NN * HD * 4);
  float* c_a  = (float*)carve((size_t)NN * CDD * 4);
  float* c_b  = (float*)carve((size_t)NN * CDD * 4);
  float* vels = (float*)carve((size_t)NN * VDD * 4);
  float* cac  = (float*)carve((size_t)NN * CDD * 4);
  float* cnt  = (float*)carve((size_t)NN * 4);
  _Float16* wE = (_Float16*)carve((size_t)NL * WTE * 2);
  _Float16* wN = (_Float16*)carve((size_t)NL * WTN * 2);

  // ---- weight conversion (f32 -> f16, transposed [col][k], padded) ----
  auto prep = [&](const float* src, _Float16* dst, int K, int Nc, int Kp, int Np) {
    int tot = Np * Kp;
    egnn_wprep<<<(tot + 255) / 256, 256, 0, stream>>>(src, dst, K, Nc, Kp, Np);
  };
  for (int l = 0; l < NL; ++l) {
    _Float16* we = wE + (size_t)l * WTE;
    prep(ew1 + (size_t)l * DIN * HD,   we,                                 DIN,    HD,  DINP,   HD);
    prep(ew2 + (size_t)l * HD * HD,    we + HD * DINP,                     HD,     HD,  HD,     HD);
    prep(cw1 + (size_t)l * HD * HD,    we + HD * DINP + HD * HD,           HD,     HD,  HD,     HD);
    prep(cw2 + (size_t)l * HD * CDD,   we + HD * DINP + 2 * HD * HD,       HD,     CDD, HD,     16);
    _Float16* wn = wN + (size_t)l * WTN;
    prep(vw1 + (size_t)l * HD * HD,    wn,                                 HD,     HD,  HD,     HD);
    prep(vw2 + (size_t)l * HD * 1,     wn + HD * HD,                       HD,     1,   HD,     16);
    prep(nw1 + (size_t)l * 2 * HD * HD, wn + HD * HD + 16 * HD,            2 * HD, HD,  2 * HD, HD);
    prep(nw2 + (size_t)l * HD * HD,    wn + HD * HD + 16 * HD + HD * 2 * HD, HD,   HD,  HD,     HD);
  }

  // ---- embedding, coords/vels extraction, segment counts ----
  egnn_embed<<<NN, HD, 0, stream>>>(nodes, emb_w, emb_b, h_a, c_a, vels);
  hipMemsetAsync(cnt, 0, (size_t)NN * 4, stream);
  egnn_cnt<<<(NE + 255) / 256, 256, 0, stream>>>(eidx, cnt);

  const size_t ldsE = (size_t)WTE * 2 + 8 * (16 * DINP + 16 * HD) * 2 +
                      8 * (16 + 16 + 48) * 4;                       // ~252 KB
  const size_t ldsN = (size_t)WTN * 2 + 8 * (16 * 2 * HD + 16 * HD) * 2 +
                      8 * 16 * 4;                                   // ~234 KB

  float* hc = h_a; float* hn = h_b;
  float* cc = c_a; float* cn = c_b;
  for (int l = 0; l < NL; ++l) {
    hipMemsetAsync(agg, 0, (size_t)NN * HD * 4, stream);
    hipMemsetAsync(cac, 0, (size_t)NN * CDD * 4, stream);
    egnn_edge<<<800, 256, ldsE, stream>>>(
        hc, cc, eidx, efeat, wE + (size_t)l * WTE,
        eb1 + l * HD, eb2 + l * HD, cb1 + l * HD, cb2 + l * CDD, agg, cac);
    float* cout = (l == NL - 1) ? (float*)d_out : cn;
    egnn_node<<<200, 256, ldsN, stream>>>(
        hc, agg, cc, cac, cnt, vels, wN + (size_t)l * WTN,
        vb1 + l * HD, vb2 + l * 1, nb1 + l * HD, nb2 + l * HD, hn, cout);
    float* tp = hc; hc = hn; hn = tp;
    tp = cc; cc = cn; cn = tp;
  }
}
